// LinearTransformer_4020089389506
// MI455X (gfx1250) — matrix operand order repaired and checked by reading, not compiled, not run on hardware
//
#include <hip/hip_runtime.h>
#include <hip/hip_bf16.h>
#include <math.h>

typedef __bf16 v8bf  __attribute__((ext_vector_type(8)));
typedef __bf16 v16bf __attribute__((ext_vector_type(16)));
typedef float  v8f   __attribute__((ext_vector_type(8)));

#define LDT 40  // LDS row stride (bf16 elements) for 32-wide K tiles (pad vs bank conflicts)

__device__ __forceinline__ v16bf cat16(v8bf lo, v8bf hi){
  v16bf r;
#pragma unroll
  for (int i = 0; i < 8; ++i){ r[i] = lo[i]; r[i+8] = hi[i]; }
  return r;
}

__device__ __forceinline__ float gelu_exact(float x){
  return 0.5f * x * (1.0f + erff(x * 0.70710678118654752440f));
}

__device__ __forceinline__ float waveSum(float v){
#pragma unroll
  for (int o = 16; o > 0; o >>= 1) v += __shfl_xor(v, o, 32);
  return v;
}
__device__ __forceinline__ float waveMax(float v){
#pragma unroll
  for (int o = 16; o > 0; o >>= 1) v = fmaxf(v, __shfl_xor(v, o, 32));
  return v;
}

// ---------------------------------------------------------------------------
// f32 (R x C) -> bf16 transpose (C x R), dims are multiples of 32 here.
// ---------------------------------------------------------------------------
__global__ __launch_bounds__(256)
void transpose_to_bf16(const float* __restrict__ in, __bf16* __restrict__ out,
                       int R, int C){
  __shared__ float tile[32][33];
  const int c0 = blockIdx.x * 32, r0 = blockIdx.y * 32;
  const int tx = threadIdx.x & 31;
  const int ty = threadIdx.x >> 5;      // 0..7
#pragma unroll
  for (int i = ty; i < 32; i += 8)
    tile[i][tx] = in[(size_t)(r0 + i) * C + (c0 + tx)];
  __syncthreads();
#pragma unroll
  for (int i = ty; i < 32; i += 8)
    out[(size_t)(c0 + i) * R + (r0 + tx)] = (__bf16)tile[tx][i];
}

// ---------------------------------------------------------------------------
// Row-wise LayerNorm (population var, eps=1e-5), bf16 output. One block/row.
// ---------------------------------------------------------------------------
__global__ __launch_bounds__(256)
void ln_to_bf16(const float* __restrict__ x, const float* __restrict__ g,
                const float* __restrict__ beta, __bf16* __restrict__ out, int D){
  const int row = blockIdx.x;
  const float* xr = x + (size_t)row * D;
  float s = 0.f, s2 = 0.f;
  for (int i = threadIdx.x; i < D; i += 256){ float v = xr[i]; s += v; s2 += v * v; }
  s  = waveSum(s);
  s2 = waveSum(s2);
  __shared__ float ps[8], ps2[8];
  __shared__ float mean_s, inv_s;
  const int wid = threadIdx.x >> 5, lane = threadIdx.x & 31;
  if (lane == 0){ ps[wid] = s; ps2[wid] = s2; }
  __syncthreads();
  if (threadIdx.x == 0){
    float a = 0.f, c = 0.f;
#pragma unroll
    for (int i = 0; i < 8; ++i){ a += ps[i]; c += ps2[i]; }
    float mean = a / (float)D;
    float var  = c / (float)D - mean * mean;
    mean_s = mean;
    inv_s  = rsqrtf(var + 1e-5f);
  }
  __syncthreads();
  const float mean = mean_s, inv = inv_s;
  for (int i = threadIdx.x; i < D; i += 256)
    out[(size_t)row * D + i] = (__bf16)((xr[i] - mean) * inv * g[i] + beta[i]);
}

// ---------------------------------------------------------------------------
// C(MxN) = A(MxK bf16 row-major) * B (given as B^T: NxK bf16 row-major)
// MODE 0: Cf = acc                         (f32)
// MODE 1: Cf = res + acc + bias[col]       (f32)
// MODE 2: Cb = bf16( gelu(acc + bias[col]) )
// Block tile 128x128, BK=32. 8 waves: wave = 32 rows x 64 cols (2x4 WMMA tiles).
// ---------------------------------------------------------------------------
template<int MODE>
__global__ __launch_bounds__(256)
void gemm_bf16(const __bf16* __restrict__ A, const __bf16* __restrict__ BT,
               const float* __restrict__ bias, const float* __restrict__ res,
               float* __restrict__ Cf, __bf16* __restrict__ Cb,
               int M, int N, int K)
{
  __shared__ alignas(32) __bf16 As[128 * LDT];
  __shared__ alignas(32) __bf16 Bs[128 * LDT];

  const int mBlk = blockIdx.y * 128;
  const int nBlk = blockIdx.x * 128;
  const int t    = threadIdx.x;
  const int lane = t & 31;
  const int wid  = t >> 5;
  const int wm   = wid & 3;        // row slice: wm*32
  const int wn   = wid >> 2;       // col slice: wn*64
  const int l    = lane & 15;
  const int half = lane >> 4;

  const int ldRow = t >> 1;        // 0..127
  const int ldCol = (t & 1) * 16;  // 0 or 16

  v8f acc[2][4] = {};

  const int kTiles = K >> 5;
  for (int kt = 0; kt < kTiles; ++kt){
    const int k0 = kt << 5;
    // cooperative global -> LDS: both tiles are K-contiguous per row
    const __bf16* ga = A  + (size_t)(mBlk + ldRow) * K + k0 + ldCol;
    const __bf16* gb = BT + (size_t)(nBlk + ldRow) * K + k0 + ldCol;
    v8bf a0 = *(const v8bf*)ga;
    v8bf a1 = *(const v8bf*)(ga + 8);
    v8bf b0 = *(const v8bf*)gb;
    v8bf b1 = *(const v8bf*)(gb + 8);
    *(v8bf*)&As[ldRow * LDT + ldCol]     = a0;
    *(v8bf*)&As[ldRow * LDT + ldCol + 8] = a1;
    *(v8bf*)&Bs[ldRow * LDT + ldCol]     = b0;
    *(v8bf*)&Bs[ldRow * LDT + ldCol + 8] = b1;
    if (kt + 1 < kTiles){                       // warm L2/L0 for next K tile
      __builtin_prefetch(ga + 32, 0, 1);
      __builtin_prefetch(gb + 32, 0, 1);
    }
    __syncthreads();

    // A frag (16x32 bf16): lanes 0-15 row m0+l holds K{0..7,16..23},
    // lanes 16-31 same rows hold K{8..15,24..31}
    v16bf af[2];
#pragma unroll
    for (int m = 0; m < 2; ++m){
      const __bf16* pa = &As[(wm * 32 + m * 16 + l) * LDT + half * 8];
      af[m] = cat16(*(const v8bf*)pa, *(const v8bf*)(pa + 16));
    }
    // B frag (32x16 bf16): lane holds one column; lanes 0-15 K{0..7,16..23},
    // lanes 16-31 K{8..15,24..31} (B^T rows are K-contiguous in LDS)
    v16bf bfr[4];
#pragma unroll
    for (int n = 0; n < 4; ++n){
      const __bf16* pb = &Bs[(wn * 64 + n * 16 + l) * LDT + half * 8];
      bfr[n] = cat16(*(const v8bf*)pb, *(const v8bf*)(pb + 16));
    }
#pragma unroll
    for (int m = 0; m < 2; ++m)
#pragma unroll
      for (int n = 0; n < 4; ++n)
        acc[m][n] = __builtin_amdgcn_wmma_f32_16x16x32_bf16(
            false, af[m], false, bfr[n], (short)0, acc[m][n], false, false);
    __syncthreads();
  }

  // Writeback: C/D layout — lanes 0-15 N=l M=v, lanes 16-31 N=l M=8+v
#pragma unroll
  for (int m = 0; m < 2; ++m){
#pragma unroll
    for (int n = 0; n < 4; ++n){
      const int col  = nBlk + wn * 64 + n * 16 + l;
      const int row0 = mBlk + wm * 32 + m * 16 + half * 8;
      float bcol = 0.f;
      if (MODE != 0) bcol = bias[col];
#pragma unroll
      for (int v = 0; v < 8; ++v){
        const size_t idx = (size_t)(row0 + v) * N + col;
        const float val = acc[m][n][v];
        if (MODE == 0)      Cf[idx] = val;
        else if (MODE == 1) Cf[idx] = res[idx] + val + bcol;
        else                Cb[idx] = (__bf16)gelu_exact(val + bcol);
      }
    }
  }
}

// ---------------------------------------------------------------------------
// Causal linear attention scan: one block per (b,h). 64x64 f32 state in LDS.
// q -> softmax(feature)*dh^-0.5, k -> exp(k); inclusive cumsums over n.
// Output written as bf16 directly into the (BN, D) layout for the O-proj GEMM.
// ---------------------------------------------------------------------------
__global__ __launch_bounds__(256)
void linattn_scan(const float* __restrict__ Q, const float* __restrict__ K,
                  const float* __restrict__ V, __bf16* __restrict__ Aout,
                  int N, int D, int H)
{
  const int dh = 64;
  const int bh = blockIdx.x;
  const int b  = bh / H;
  const int h  = bh % H;
  __shared__ float S[64 * 65];
  __shared__ float zc[64];
  __shared__ float qv[64], kv[64], vv[64];
  __shared__ float pout[256];
  __shared__ float dinv_s;
  const int t = threadIdx.x;

  for (int i = t; i < 64 * 65; i += 256) S[i] = 0.f;
  if (t < 64) zc[t] = 0.f;
  __syncthreads();

  const size_t base = (size_t)b * N * D + (size_t)h * dh;
  for (int n = 0; n < N; ++n){
    const size_t off = base + (size_t)n * D;
    if (t < 64){
      qv[t] = Q[off + t];
      kv[t] = __expf(K[off + t]);
      vv[t] = V[off + t];
    }
    __syncthreads();
    if (t < 32){  // wave 0: softmax(q), inclusive z cumsum, denominator
      float a0 = qv[t], a1 = qv[t + 32];
      float mx = waveMax(fmaxf(a0, a1));
      float e0 = __expf(a0 - mx), e1 = __expf(a1 - mx);
      float sm = waveSum(e0 + e1);
      const float sc = 0.125f / sm;        // * dh^-0.5 (dh=64)
      e0 *= sc; e1 *= sc;
      qv[t] = e0; qv[t + 32] = e1;
      float z0 = zc[t] + kv[t];
      float z1 = zc[t + 32] + kv[t + 32];
      zc[t] = z0; zc[t + 32] = z1;
      float den = waveSum(e0 * z0 + e1 * z1);
      if (t == 0) dinv_s = 1.0f / fmaxf(den, 1e-3f);
    }
    __syncthreads();
    {   // S += k v^T (inclusive), partial column dot q^T S
      const int jj = t & 63;
      const int ib = t >> 6;               // 0..3 -> 16-row strip
      const float vj = vv[jj];
      float p = 0.f;
      const int i0 = ib * 16;
#pragma unroll
      for (int ii = 0; ii < 16; ++ii){
        const int i = i0 + ii;
        const float s = S[i * 65 + jj] + kv[i] * vj;
        S[i * 65 + jj] = s;
        p += qv[i] * s;
      }
      pout[t] = p;
    }
    __syncthreads();
    if (t < 64){
      const float o = (pout[t] + pout[t + 64] + pout[t + 128] + pout[t + 192]) * dinv_s;
      Aout[off + t] = (__bf16)o;
    }
    __syncthreads();
  }
}

// ---------------------------------------------------------------------------
extern "C" void kernel_launch(void* const* d_in, const int* in_sizes, int n_in,
                              void* d_out, int out_size, void* d_ws, size_t ws_size,
                              hipStream_t stream)
{
  (void)in_sizes; (void)n_in; (void)out_size; (void)ws_size;
  const int Bb = 2, N = 2048, D = 1024, H = 16, DEPTH = 2, FF = 4096;
  const int BN = Bb * N;

  float* x = (float*)d_out;   // residual stream lives in the output buffer
  hipMemcpyAsync(x, d_in[0], sizeof(float) * (size_t)BN * D,
                 hipMemcpyDeviceToDevice, stream);

  char* ws = (char*)d_ws;
  size_t off = 0;
  auto take = [&](size_t bytes) -> char* {
    char* p = ws + off;
    off += (bytes + 255) & ~(size_t)255;
    return p;
  };
  __bf16* wtq = (__bf16*)take((size_t)D * D * 2);
  __bf16* wtk = (__bf16*)take((size_t)D * D * 2);
  __bf16* wtv = (__bf16*)take((size_t)D * D * 2);
  __bf16* wto = (__bf16*)take((size_t)D * D * 2);
  __bf16* wt1 = (__bf16*)take((size_t)D * FF * 2);
  __bf16* wt2 = (__bf16*)take((size_t)FF * D * 2);
  __bf16* xn  = (__bf16*)take((size_t)BN * D * 2);
  float*  qb  = (float*) take((size_t)BN * D * 4);
  float*  kb  = (float*) take((size_t)BN * D * 4);
  float*  vb  = (float*) take((size_t)BN * D * 4);
  __bf16* ab  = (__bf16*)take((size_t)BN * D * 2);
  __bf16* hb  = (__bf16*)qb;   // FF hidden (32MB) aliases dead Q+K region

  const dim3 blk(256);
  const dim3 gProj(D / 128, BN / 128);     // N=1024 outputs
  const dim3 gFF1(FF / 128, BN / 128);     // N=4096 outputs

  for (int L = 0; L < DEPTH; ++L){
    const float* Wq  = (const float*)d_in[3]  + (size_t)L * D * D;
    const float* Wk  = (const float*)d_in[4]  + (size_t)L * D * D;
    const float* Wv  = (const float*)d_in[5]  + (size_t)L * D * D;
    const float* Wo  = (const float*)d_in[6]  + (size_t)L * D * D;
    const float* W1  = (const float*)d_in[10] + (size_t)L * D * FF;
    const float* W2  = (const float*)d_in[12] + (size_t)L * FF * D;
    const float* g1  = (const float*)d_in[1]  + (size_t)L * D;
    const float* be1 = (const float*)d_in[2]  + (size_t)L * D;
    const float* bo  = (const float*)d_in[7]  + (size_t)L * D;
    const float* g2  = (const float*)d_in[8]  + (size_t)L * D;
    const float* be2 = (const float*)d_in[9]  + (size_t)L * D;
    const float* bf1 = (const float*)d_in[11] + (size_t)L * FF;
    const float* bf2 = (const float*)d_in[13] + (size_t)L * D;

    // weight -> bf16 W^T (NxK) so both GEMM operands are K-contiguous
    transpose_to_bf16<<<dim3(D / 32, D / 32), blk, 0, stream>>>(Wq, wtq, D, D);
    transpose_to_bf16<<<dim3(D / 32, D / 32), blk, 0, stream>>>(Wk, wtk, D, D);
    transpose_to_bf16<<<dim3(D / 32, D / 32), blk, 0, stream>>>(Wv, wtv, D, D);
    transpose_to_bf16<<<dim3(D / 32, D / 32), blk, 0, stream>>>(Wo, wto, D, D);
    transpose_to_bf16<<<dim3(FF / 32, D / 32), blk, 0, stream>>>(W1, wt1, D, FF);
    transpose_to_bf16<<<dim3(D / 32, FF / 32), blk, 0, stream>>>(W2, wt2, FF, D);

    // attention block
    ln_to_bf16<<<BN, blk, 0, stream>>>(x, g1, be1, xn, D);
    gemm_bf16<0><<<gProj, blk, 0, stream>>>(xn, wtq, nullptr, nullptr, qb, nullptr, BN, D, D);
    gemm_bf16<0><<<gProj, blk, 0, stream>>>(xn, wtk, nullptr, nullptr, kb, nullptr, BN, D, D);
    gemm_bf16<0><<<gProj, blk, 0, stream>>>(xn, wtv, nullptr, nullptr, vb, nullptr, BN, D, D);
    linattn_scan<<<Bb * H, blk, 0, stream>>>(qb, kb, vb, ab, N, D, H);
    gemm_bf16<1><<<gProj, blk, 0, stream>>>(ab, wto, bo, x, x, nullptr, BN, D, D);

    // feed-forward block
    ln_to_bf16<<<BN, blk, 0, stream>>>(x, g2, be2, xn, D);
    gemm_bf16<2><<<gFF1, blk, 0, stream>>>(xn, wt1, bf1, nullptr, nullptr, hb, BN, FF, D);
    gemm_bf16<1><<<gProj, blk, 0, stream>>>(hb, wt2, bf2, x, x, nullptr, BN, D, FF);
  }
}
